// Qwen3VLVisionEncoder_85246510891399
// MI455X (gfx1250) — matrix operand order repaired and checked by reading, not compiled, not run on hardware
//
#include <hip/hip_runtime.h>
#include <hip/hip_bf16.h>
#include <math.h>

// ---------------------------------------------------------------------------
// Model constants
// ---------------------------------------------------------------------------
enum {
  B_ = 2, S_ = 1024, D_ = 1280, H_ = 16, HD_ = 80, F_ = 5120, L_ = 4,
  CIN_ = 1176, CINP_ = 1184,            // CIN padded to multiple of 32
  HDP_ = 96,                            // head dim padded to multiple of 32
  NP_ = 2304, OUT_ = 2048, GW_ = 32,
  MTOK_ = B_ * S_                       // 2048 token rows
};

typedef __attribute__((ext_vector_type(16))) __bf16 v16bf;
typedef __attribute__((ext_vector_type(8)))  float  v8f;

static __device__ __forceinline__ float gelu_exact(float x) {
  return 0.5f * x * (1.0f + erff(x * 0.70710678118654752440f));
}

// ---------------------------------------------------------------------------
// f32 -> bf16 conversion with K padding (pad region zero-filled)
// ---------------------------------------------------------------------------
__global__ void cvt_pad_bf16_kernel(const float* __restrict__ src,
                                    __bf16* __restrict__ dst,
                                    int K, int Kp, long total) {
  long idx = (long)blockIdx.x * blockDim.x + threadIdx.x;
  if (idx >= total) return;
  long r = idx / Kp;
  int  c = (int)(idx - r * Kp);
  dst[idx] = (c < K) ? (__bf16)src[r * (long)K + c] : (__bf16)0.0f;
}

// ---------------------------------------------------------------------------
// x[m,n] += pos_table[pos_ids[m], n]
// ---------------------------------------------------------------------------
__global__ void add_pos_kernel(float* __restrict__ x,
                               const float* __restrict__ pos_table,
                               const int* __restrict__ pos_ids, long total) {
  long idx = (long)blockIdx.x * blockDim.x + threadIdx.x;
  if (idx >= total) return;
  long m = idx / D_;
  int  n = (int)(idx - m * D_);
  x[idx] += pos_table[(long)pos_ids[m] * D_ + n];
}

// ---------------------------------------------------------------------------
// LayerNorm over D_=1280, one row per workgroup, bf16 output
// ---------------------------------------------------------------------------
__global__ void __launch_bounds__(256)
layernorm_bf16_kernel(const float* __restrict__ x,
                      const float* __restrict__ w, const float* __restrict__ b,
                      __bf16* __restrict__ out) {
  __shared__ float r1[256], r2[256];
  long row = blockIdx.x;
  const float* xr = x + row * (long)D_;
  int tid = threadIdx.x;
  float s = 0.f, s2 = 0.f;
  for (int i = tid; i < D_; i += 256) { float v = xr[i]; s += v; s2 += v * v; }
  r1[tid] = s; r2[tid] = s2; __syncthreads();
  for (int off = 128; off > 0; off >>= 1) {
    if (tid < off) { r1[tid] += r1[tid + off]; r2[tid] += r2[tid + off]; }
    __syncthreads();
  }
  float mean = r1[0] * (1.0f / D_);
  float var  = r2[0] * (1.0f / D_) - mean * mean;
  float rstd = rsqrtf(var + 1e-6f);
  __bf16* orow = out + row * (long)D_;
  for (int i = tid; i < D_; i += 256)
    orow[i] = (__bf16)((xr[i] - mean) * rstd * w[i] + b[i]);
}

// ---------------------------------------------------------------------------
// Fused RoPE + head repack.
//  qkv   : f32 [B,S,3,H,HD]
//  q_bf  : bf16 [B,H,S,HDP_]  (cols 80..95 zero)
//  k_bf  : bf16 [B,H,S,HDP_]
//  vT_bf : bf16 [B,H,HD,S]    (transposed so p@v is an A@W^T GEMM)
// ---------------------------------------------------------------------------
__global__ void __launch_bounds__(640)
rope_repack_kernel(const float* __restrict__ qkv,
                   __bf16* __restrict__ q_bf, __bf16* __restrict__ k_bf,
                   __bf16* __restrict__ vT_bf) {
  int bs = blockIdx.x;                 // b*S + s
  int b  = bs >> 10;
  int s  = bs & (S_ - 1);
  int t  = threadIdx.x;
  int h  = t / 40;
  int i  = t - h * 40;                 // 0..39
  int ii = (i < 20) ? i : i - 20;
  float posf = (i < 20) ? (float)(s / GW_) : (float)(s % GW_);
  float inv  = __expf(-__logf(10000.0f) * ((float)(2 * ii) / 40.0f));
  float th   = posf * inv;
  float c = __cosf(th), sn = __sinf(th);

  const float* base = qkv + (long)bs * (3 * D_);
  const float* qh = base + h * HD_;
  const float* kh = base + D_ + h * HD_;
  const float* vh = base + 2 * D_ + h * HD_;
  float q0 = qh[i], q1 = qh[i + 40];
  float k0 = kh[i], k1 = kh[i + 40];
  float v0 = vh[i], v1 = vh[i + 40];

  long bh  = (long)(b * H_ + h);
  long qkb = (bh * S_ + s) * HDP_;
  q_bf[qkb + i]      = (__bf16)(q0 * c - q1 * sn);
  q_bf[qkb + i + 40] = (__bf16)(q1 * c + q0 * sn);
  k_bf[qkb + i]      = (__bf16)(k0 * c - k1 * sn);
  k_bf[qkb + i + 40] = (__bf16)(k1 * c + k0 * sn);
  if (i < HDP_ - HD_) {                // zero the K-pad (cols 80..95)
    q_bf[qkb + HD_ + i] = (__bf16)0.0f;
    k_bf[qkb + HD_ + i] = (__bf16)0.0f;
  }
  long vtb = (bh * HD_ + i) * S_ + s;
  vT_bf[vtb]            = (__bf16)v0;
  vT_bf[vtb + 40L * S_] = (__bf16)v1;
}

// ---------------------------------------------------------------------------
// Masked softmax over the key axis; writes bf16 probabilities.
// Invalid query rows are zeroed entirely (matches reference o*valid).
// ---------------------------------------------------------------------------
__global__ void __launch_bounds__(256)
softmax_mask_kernel(const float* __restrict__ scores,
                    __bf16* __restrict__ p,
                    const int* __restrict__ seq_lens) {
  __shared__ float red[256];
  long row = blockIdx.x;               // (b*H + h)*S + q
  int b = (int)(row / ((long)H_ * S_));
  int q = (int)(row & (S_ - 1));
  int len = seq_lens[b];
  const float* sr = scores + row * (long)S_;
  __bf16* pr = p + row * (long)S_;
  int tid = threadIdx.x;

  if (q >= len) {
    for (int k = tid; k < S_; k += 256) pr[k] = (__bf16)0.0f;
    return;
  }
  float mx = -3.4e38f;
  for (int k = tid; k < len; k += 256) mx = fmaxf(mx, sr[k]);
  red[tid] = mx; __syncthreads();
  for (int off = 128; off > 0; off >>= 1) {
    if (tid < off) red[tid] = fmaxf(red[tid], red[tid + off]);
    __syncthreads();
  }
  mx = red[0]; __syncthreads();
  float sum = 0.f;
  for (int k = tid; k < len; k += 256) sum += __expf(sr[k] - mx);
  red[tid] = sum; __syncthreads();
  for (int off = 128; off > 0; off >>= 1) {
    if (tid < off) red[tid] += red[tid + off];
    __syncthreads();
  }
  float rs = 1.0f / red[0];
  for (int k = tid; k < S_; k += 256)
    pr[k] = (k < len) ? (__bf16)(__expf(sr[k] - mx) * rs) : (__bf16)0.0f;
}

// ---------------------------------------------------------------------------
// Repack attention output: o_f [B,H,S,HD] f32 -> o_bf [B,S,H*HD] bf16
// ---------------------------------------------------------------------------
__global__ void repack_o_kernel(const float* __restrict__ o_f,
                                __bf16* __restrict__ o_bf, long total) {
  long idx = (long)blockIdx.x * blockDim.x + threadIdx.x;
  if (idx >= total) return;
  long m = idx / D_;                   // b*S + s
  int  n = (int)(idx - m * D_);        // h*HD + d
  int  b = (int)(m >> 10);
  int  s = (int)(m & (S_ - 1));
  int  h = n / HD_;
  int  d = n - h * HD_;
  o_bf[idx] = (__bf16)o_f[(((long)(b * H_ + h) * S_ + s) * HD_) + d];
}

// ---------------------------------------------------------------------------
// Tiled bf16 WMMA GEMM:  out[z][M,N] = alpha * A[z][M,K] @ W[z][N,K]^T
//                                      (+bias[N]) (+res[M,N]) (GELU?)
// 256 threads (8 waves), 128x128 tile, BK=32; wave tile 64x32 (4x2 WMMAs).
// Interior blocks: double-buffered GLOBAL_LOAD_ASYNC_TO_LDS_B128 staging
//   (ASYNCcnt-tracked, next tile's fetch overlaps current tile's WMMAs).
// Edge blocks (only the N=80 attention o-GEMM here): guarded sync staging.
// K must be a multiple of 32.
// ---------------------------------------------------------------------------
#define LDS_STRIDE 40                  // 32 + 8 halves pad -> conflict-free

union FragBF { v16bf v; unsigned u[8]; };

static __device__ __forceinline__ unsigned lds_off32(const void* p) {
  // Generic pointers to LDS carry the LDS byte offset in the low 32 bits.
  return (unsigned)(size_t)p;
}

template <bool BIAS, bool RES, bool GELU, bool OBF>
__global__ void __launch_bounds__(256)
gemm_bf16_kernel(const __bf16* __restrict__ A, const __bf16* __restrict__ W,
                 const float* __restrict__ bias, const float* __restrict__ res,
                 void* __restrict__ out, int M, int N, int K,
                 long batchA, long batchW, long batchOut, float alpha) {
  __shared__ __align__(16) unsigned short sA[2][128 * LDS_STRIDE];
  __shared__ __align__(16) unsigned short sB[2][128 * LDS_STRIDE];

  int z = blockIdx.z;
  A += (long)z * batchA;
  W += (long)z * batchW;

  int mBase = blockIdx.y * 128;
  int nBase = blockIdx.x * 128;
  int tid  = threadIdx.x;
  int wave = tid >> 5;
  int lane = tid & 31;
  int wm   = wave >> 2;                // 0..1 -> 64-row slab
  int wn   = wave & 3;                 // 0..3 -> 32-col slab
  int hl   = lane >> 4;                // half-wave
  int lm   = lane & 15;

  v8f acc[4][2] = {};
  int ksteps = K >> 5;
  bool interior = (mBase + 128 <= M) && (nBase + 128 <= N);

  // per-thread staging coordinates (2 chunks of 16B per matrix per tile)
  int r0  = tid >> 2;                  // chunk 0 row
  int c80 = (tid & 3) << 3;
  int r1  = (tid + 256) >> 2;          // chunk 1 row
  int c81 = ((tid + 256) & 3) << 3;

  if (interior) {
    // -------------------- async double-buffered path --------------------
    auto issue_tile = [&](int kt, int buf) {
      int k0 = kt << 5;
      const __bf16* ga0 = A + (long)(mBase + r0) * K + k0 + c80;
      unsigned la0 = lds_off32(&sA[buf][r0 * LDS_STRIDE + c80]);
      asm volatile("global_load_async_to_lds_b128 %0, %1, off"
                   :: "v"(la0), "v"(ga0) : "memory");
      const __bf16* gb0 = W + (long)(nBase + r0) * K + k0 + c80;
      unsigned lb0 = lds_off32(&sB[buf][r0 * LDS_STRIDE + c80]);
      asm volatile("global_load_async_to_lds_b128 %0, %1, off"
                   :: "v"(lb0), "v"(gb0) : "memory");
      const __bf16* ga1 = A + (long)(mBase + r1) * K + k0 + c81;
      unsigned la1 = lds_off32(&sA[buf][r1 * LDS_STRIDE + c81]);
      asm volatile("global_load_async_to_lds_b128 %0, %1, off"
                   :: "v"(la1), "v"(ga1) : "memory");
      const __bf16* gb1 = W + (long)(nBase + r1) * K + k0 + c81;
      unsigned lb1 = lds_off32(&sB[buf][r1 * LDS_STRIDE + c81]);
      asm volatile("global_load_async_to_lds_b128 %0, %1, off"
                   :: "v"(lb1), "v"(gb1) : "memory");
    };

    issue_tile(0, 0);
    for (int kt = 0; kt < ksteps; ++kt) {
      int cur = kt & 1;
      __syncthreads();                 // everyone done reading buf[cur^1]
      if (kt + 1 < ksteps) {
        issue_tile(kt + 1, cur ^ 1);   // overlap with this tile's compute
        asm volatile("s_wait_asynccnt 0x4" ::: "memory");  // tile kt landed
      } else {
        asm volatile("s_wait_asynccnt 0x0" ::: "memory");
      }
      __syncthreads();                 // all waves' tile-kt data visible

      FragBF a[4], b[2];
      #pragma unroll
      for (int tm = 0; tm < 4; ++tm) {
        int m = wm * 64 + tm * 16 + lm;
        #pragma unroll
        for (int v = 0; v < 8; ++v) {
          int g = v >> 2, j = v & 3;
          int k = hl * 16 + 2 * v;
          a[tm].u[v] = *(const unsigned*)&sA[cur][m * LDS_STRIDE + k];
        }
      }
      #pragma unroll
      for (int tn = 0; tn < 2; ++tn) {
        int n = wn * 32 + tn * 16 + lm;
        #pragma unroll
        for (int v = 0; v < 8; ++v) {
          int k = hl * 16 + 2 * v;
          b[tn].u[v] = *(const unsigned*)&sB[cur][n * LDS_STRIDE + k];
        }
      }
      #pragma unroll
      for (int tm = 0; tm < 4; ++tm)
        #pragma unroll
        for (int tn = 0; tn < 2; ++tn)
          acc[tm][tn] = __builtin_amdgcn_wmma_f32_16x16x32_bf16(
              false, a[tm].v, false, b[tn].v, (short)0, acc[tm][tn],
              false, false);
    }
  } else {
    // -------------------- guarded synchronous path ----------------------
    for (int kt = 0; kt < ksteps; ++kt) {
      int k0 = kt << 5;
      __syncthreads();
      #pragma unroll
      for (int it = 0; it < 2; ++it) {
        int chunk = tid + it * 256;
        int r  = chunk >> 2;
        int c8 = (chunk & 3) << 3;
        int gm = mBase + r;
        uint4 va = {0u, 0u, 0u, 0u};
        if (gm < M) va = *(const uint4*)(A + (long)gm * K + k0 + c8);
        *(uint4*)&sA[0][r * LDS_STRIDE + c8] = va;
        int gn = nBase + r;
        uint4 vb = {0u, 0u, 0u, 0u};
        if (gn < N) vb = *(const uint4*)(W + (long)gn * K + k0 + c8);
        *(uint4*)&sB[0][r * LDS_STRIDE + c8] = vb;
      }
      __syncthreads();

      FragBF a[4], b[2];
      #pragma unroll
      for (int tm = 0; tm < 4; ++tm) {
        int m = wm * 64 + tm * 16 + lm;
        #pragma unroll
        for (int v = 0; v < 8; ++v) {
          int g = v >> 2, j = v & 3;
          int k = hl * 16 + 2 * v;
          a[tm].u[v] = *(const unsigned*)&sA[0][m * LDS_STRIDE + k];
        }
      }
      #pragma unroll
      for (int tn = 0; tn < 2; ++tn) {
        int n = wn * 32 + tn * 16 + lm;
        #pragma unroll
        for (int v = 0; v < 8; ++v) {
          int k = hl * 16 + 2 * v;
          b[tn].u[v] = *(const unsigned*)&sB[0][n * LDS_STRIDE + k];
        }
      }
      #pragma unroll
      for (int tm = 0; tm < 4; ++tm)
        #pragma unroll
        for (int tn = 0; tn < 2; ++tn)
          acc[tm][tn] = __builtin_amdgcn_wmma_f32_16x16x32_bf16(
              false, a[tm].v, false, b[tn].v, (short)0, acc[tm][tn],
              false, false);
    }
  }

  // ------------------------------ epilogue ------------------------------
  float* of  = (float*)out + (long)z * batchOut;
  __bf16* ob = (__bf16*)out + (long)z * batchOut;
  #pragma unroll
  for (int tm = 0; tm < 4; ++tm) {
    #pragma unroll
    for (int tn = 0; tn < 2; ++tn) {
      int n = nBase + wn * 32 + tn * 16 + lm;
      #pragma unroll
      for (int r = 0; r < 8; ++r) {
        int m = mBase + wm * 64 + tm * 16 + hl * 8 + r;
        if (interior || (m < M && n < N)) {
          float v = acc[tm][tn][r] * alpha;
          if constexpr (BIAS) v += bias[n];
          if constexpr (RES)  v += res[(long)m * N + n];
          if constexpr (GELU) v = gelu_exact(v);
          if constexpr (OBF)  ob[(long)m * N + n] = (__bf16)v;
          else                of[(long)m * N + n] = v;
        }
      }
    }
  }
}

// ---------------------------------------------------------------------------
// Host orchestration
// ---------------------------------------------------------------------------
static inline long cdiv(long a, long b) { return (a + b - 1) / b; }

struct Bump {
  char* base; size_t cur;
  void* take(size_t bytes) {
    void* p = base + cur;
    cur += (bytes + 255) & ~(size_t)255;
    return p;
  }
};

template <bool BIAS, bool RES, bool GELU, bool OBF>
static void launch_gemm(hipStream_t st, const __bf16* A, const __bf16* W,
                        const float* bias, const float* res, void* out,
                        int M, int N, int K, int Z,
                        long bA, long bW, long bO, float alpha) {
  dim3 grid((unsigned)cdiv(N, 128), (unsigned)cdiv(M, 128), (unsigned)Z);
  gemm_bf16_kernel<BIAS, RES, GELU, OBF><<<grid, 256, 0, st>>>(
      A, W, bias, res, out, M, N, K, bA, bW, bO, alpha);
}

static void launch_cvt(hipStream_t st, const float* src, __bf16* dst,
                       long R, int K, int Kp) {
  long total = R * Kp;
  cvt_pad_bf16_kernel<<<(unsigned)cdiv(total, 256), 256, 0, st>>>(
      src, dst, K, Kp, total);
}

extern "C" void kernel_launch(void* const* d_in, const int* in_sizes, int n_in,
                              void* d_out, int out_size, void* d_ws,
                              size_t ws_size, hipStream_t stream) {
  (void)in_sizes; (void)n_in; (void)out_size; (void)ws_size;
  const float* pixels   = (const float*)d_in[0];
  const float* patch_w  = (const float*)d_in[1];
  const float* patch_b  = (const float*)d_in[2];
  const float* pos_tab  = (const float*)d_in[3];
  const float* ln1_w    = (const float*)d_in[4];
  const float* ln1_b    = (const float*)d_in[5];
  const float* qkv_w    = (const float*)d_in[6];
  const float* qkv_b    = (const float*)d_in[7];
  const float* proj_w   = (const float*)d_in[8];
  const float* proj_b   = (const float*)d_in[9];
  const float* ln2_w    = (const float*)d_in[10];
  const float* ln2_b    = (const float*)d_in[11];
  const float* fc1_w    = (const float*)d_in[12];
  const float* fc1_b    = (const float*)d_in[13];
  const float* fc2_w    = (const float*)d_in[14];
  const float* fc2_b    = (const float*)d_in[15];
  const float* mnorm_w  = (const float*)d_in[16];
  const float* mnorm_b  = (const float*)d_in[17];
  const float* mfc1_w   = (const float*)d_in[18];
  const float* mfc1_b   = (const float*)d_in[19];
  const float* mfc2_w   = (const float*)d_in[20];
  const float* mfc2_b   = (const float*)d_in[21];
  const int*   seq_lens = (const int*)d_in[22];
  const int*   pos_ids  = (const int*)d_in[23];

  Bump ws{(char*)d_ws, 0};
  // --- bf16 weight copies ---
  __bf16* w_patch = (__bf16*)ws.take((size_t)D_ * CINP_ * 2);
  __bf16* w_qkv   = (__bf16*)ws.take((size_t)L_ * 3 * D_ * D_ * 2);
  __bf16* w_proj  = (__bf16*)ws.take((size_t)L_ * D_ * D_ * 2);
  __bf16* w_fc1   = (__bf16*)ws.take((size_t)L_ * F_ * D_ * 2);
  __bf16* w_fc2   = (__bf16*)ws.take((size_t)L_ * D_ * F_ * 2);
  __bf16* w_mfc1  = (__bf16*)ws.take((size_t)4 * D_ * 4 * D_ * 2);
  __bf16* w_mfc2  = (__bf16*)ws.take((size_t)OUT_ * 4 * D_ * 2);
  // --- activations ---
  __bf16* pix_bf  = (__bf16*)ws.take((size_t)MTOK_ * CINP_ * 2);
  float*  x_a     = (float*)ws.take((size_t)MTOK_ * D_ * 4);
  float*  x_b     = (float*)ws.take((size_t)MTOK_ * D_ * 4);
  __bf16* h_bf    = (__bf16*)ws.take((size_t)MTOK_ * D_ * 2);
  float*  qkv_f   = (float*)ws.take((size_t)MTOK_ * 3 * D_ * 4);
  __bf16* q_bf    = (__bf16*)ws.take((size_t)B_ * H_ * S_ * HDP_ * 2);
  __bf16* k_bf    = (__bf16*)ws.take((size_t)B_ * H_ * S_ * HDP_ * 2);
  __bf16* vT_bf   = (__bf16*)ws.take((size_t)B_ * H_ * HD_ * S_ * 2);
  float*  sc_f    = (float*)ws.take((size_t)B_ * H_ * S_ * S_ * 4);
  __bf16* p_bf    = (__bf16*)ws.take((size_t)B_ * H_ * S_ * S_ * 2);
  float*  o_f     = (float*)ws.take((size_t)B_ * H_ * S_ * HD_ * 4);
  __bf16* o_bf    = (__bf16*)ws.take((size_t)MTOK_ * D_ * 2);
  __bf16* ffn_bf  = (__bf16*)ws.take((size_t)MTOK_ * F_ * 2);

  // --- convert weights + pixels to bf16 (pad K where needed) ---
  launch_cvt(stream, patch_w, w_patch, D_, CIN_, CINP_);
  launch_cvt(stream, qkv_w,  w_qkv,  (long)L_ * 3 * D_, D_, D_);
  launch_cvt(stream, proj_w, w_proj, (long)L_ * D_,     D_, D_);
  launch_cvt(stream, fc1_w,  w_fc1,  (long)L_ * F_,     D_, D_);
  launch_cvt(stream, fc2_w,  w_fc2,  (long)L_ * D_,     F_, F_);
  launch_cvt(stream, mfc1_w, w_mfc1, (long)4 * D_,  4 * D_, 4 * D_);
  launch_cvt(stream, mfc2_w, w_mfc2, (long)OUT_,    4 * D_, 4 * D_);
  launch_cvt(stream, pixels, pix_bf, (long)MTOK_, CIN_, CINP_);

  // --- patch embed: x = pixels @ patch_w^T + patch_b ; x += pos_table[ids]
  launch_gemm<true, false, false, false>(stream, pix_bf, w_patch, patch_b,
      nullptr, x_a, MTOK_, D_, CINP_, 1, 0, 0, 0, 1.0f);
  add_pos_kernel<<<(unsigned)cdiv((long)MTOK_ * D_, 256), 256, 0, stream>>>(
      x_a, pos_tab, pos_ids, (long)MTOK_ * D_);

  const float attn_scale = 1.0f / sqrtf((float)HD_);
  const int ZBH = B_ * H_;

  float* xc = x_a;   // current
  float* xn = x_b;   // next
  for (int l = 0; l < L_; ++l) {
    layernorm_bf16_kernel<<<MTOK_, 256, 0, stream>>>(
        xc, ln1_w + (long)l * D_, ln1_b + (long)l * D_, h_bf);
    launch_gemm<true, false, false, false>(stream, h_bf,
        w_qkv + (long)l * 3 * D_ * D_, qkv_b + (long)l * 3 * D_, nullptr,
        qkv_f, MTOK_, 3 * D_, D_, 1, 0, 0, 0, 1.0f);
    rope_repack_kernel<<<MTOK_, H_ * 40, 0, stream>>>(qkv_f, q_bf, k_bf, vT_bf);
    launch_gemm<false, false, false, false>(stream, q_bf, k_bf, nullptr,
        nullptr, sc_f, S_, S_, HDP_, ZBH,
        (long)S_ * HDP_, (long)S_ * HDP_, (long)S_ * S_, attn_scale);
    softmax_mask_kernel<<<(unsigned)((long)ZBH * S_), 256, 0, stream>>>(
        sc_f, p_bf, seq_lens);
    launch_gemm<false, false, false, false>(stream, p_bf, vT_bf, nullptr,
        nullptr, o_f, S_, HD_, S_, ZBH,
        (long)S_ * S_, (long)HD_ * S_, (long)S_ * HD_, 1.0f);
    repack_o_kernel<<<(unsigned)cdiv((long)MTOK_ * D_, 256), 256, 0, stream>>>(
        o_f, o_bf, (long)MTOK_ * D_);
    launch_gemm<true, true, false, false>(stream, o_bf,
        w_proj + (long)l * D_ * D_, proj_b + (long)l * D_, xc, xn,
        MTOK_, D_, D_, 1, 0, 0, 0, 1.0f);
    layernorm_bf16_kernel<<<MTOK_, 256, 0, stream>>>(
        xn, ln2_w + (long)l * D_, ln2_b + (long)l * D_, h_bf);
    launch_gemm<true, false, true, true>(stream, h_bf,
        w_fc1 + (long)l * F_ * D_, fc1_b + (long)l * F_, nullptr, ffn_bf,
        MTOK_, F_, D_, 1, 0, 0, 0, 1.0f);
    launch_gemm<true, true, false, false>(stream, ffn_bf,
        w_fc2 + (long)l * D_ * F_, fc2_b + (long)l * D_, xn, xc,
        MTOK_, D_, F_, 1, 0, 0, 0, 1.0f);
  }

  // --- merger: LN over D, reinterpret [2048,1280]bf16 as [512,5120] ---
  layernorm_bf16_kernel<<<MTOK_, 256, 0, stream>>>(xc, mnorm_w, mnorm_b, h_bf);
  const int MM = MTOK_ / 4;            // 512
  launch_gemm<true, false, true, true>(stream, h_bf, w_mfc1, mfc1_b, nullptr,
      ffn_bf, MM, 4 * D_, 4 * D_, 1, 0, 0, 0, 1.0f);
  launch_gemm<true, false, false, false>(stream, ffn_bf, w_mfc2, mfc2_b,
      nullptr, (float*)d_out, MM, OUT_, 4 * D_, 1, 0, 0, 0, 1.0f);
}
